// EpochedFutureFill_47820165873725
// MI455X (gfx1250) — matrix operand order repaired and checked by reading, not compiled, not run on hardware
//
#include <hip/hip_runtime.h>

// Causal convolution y[b,t] = sum_s filt[s] * x[b,t-s]  (B=256, T=L=32768)
// as a Toeplitz GEMM on CDNA5 WMMA pipes. bf16-split (hi+lo) operands,
// fp32 accumulation: 3 x v_wmma_f32_16x16x32_bf16 per 16x16x32 tile.
//
//   prep: x -> bf16 hi/lo planes; filt -> 8 shift-copies of the REVERSED,
//         zero-padded bf16 filter (hi/lo). Shift copies make every LDS
//         B-fragment read 16B-aligned (bb mod 8 depends only on the lane).
//   GEMM: block = 16 batches x 256 time; 8 waves x 2 N-tiles; K-step 64;
//         double-buffered async-to-LDS staging (global_load_async_to_lds_b128,
//         aligned 16B chunks), software-pipelined: issue tile i+1, then run
//         12 WMMAs on tile i. One barrier per iteration.

typedef __attribute__((ext_vector_type(16))) __bf16 v16bf;
typedef __attribute__((ext_vector_type(8)))  float  v8f;
typedef unsigned short u16_t;
typedef unsigned int   u32_t;

#define USE_ASYNC_STAGE 1
#define XP  72      // LDS row stride for x tiles (bf16 elems)
#define BN  256     // block N (time) span
#define BK  64      // lags staged per iteration
#define FRW 320     // filter window entries staged per copy per iteration
#define FCP 336     // LDS stride per filter copy-plane (bf16 elems, 672B)
#define STG (32 * XP + 16 * FCP)   // elems per stage buffer (x hi/lo + 16 copy-planes)

union FragB16 { v16bf v; u32_t u[8]; u16_t s[16]; };

__device__ __forceinline__ u16_t bf16_rne(u32_t u) {
    return (u16_t)((u + 0x7FFFu + ((u >> 16) & 1u)) >> 16);
}
__device__ __forceinline__ void split_bf16(float v, u16_t& hi, u16_t& lo) {
    u32_t u = __float_as_uint(v);
    hi = bf16_rne(u);
    float fhi = __uint_as_float((u32_t)hi << 16);
    lo = bf16_rne(__float_as_uint(v - fhi));
}

__device__ __forceinline__ void ld2q(FragB16& f, const u16_t* p0, const u16_t* p1) {
    const uint4 q0 = *(const uint4*)p0;   // 16B-aligned by construction
    const uint4 q1 = *(const uint4*)p1;
    f.u[0] = q0.x; f.u[1] = q0.y; f.u[2] = q0.z; f.u[3] = q0.w;
    f.u[4] = q1.x; f.u[5] = q1.y; f.u[6] = q1.z; f.u[7] = q1.w;
}

// ---------------- prep kernels ----------------------------------------------

__global__ void __launch_bounds__(256)
prep_split_x(const float* __restrict__ x, u16_t* __restrict__ xh,
             u16_t* __restrict__ xl, int n) {
    int i = blockIdx.x * 256 + threadIdx.x;
    if (i < n) {
        u16_t h, l;
        split_bf16(x[i], h, l);
        xh[i] = h; xl[i] = l;
    }
}

// copy-plane sp = s*2 + plane (s = shift 0..7, plane 0=hi,1=lo):
//   frws[sp*FS + i] = bf16plane( filt[L-1-s-i] ), zero outside [0,L)
__global__ void __launch_bounds__(256)
prep_split_filt_copies(const float* __restrict__ filt, u16_t* __restrict__ frws,
                       int L, int FS) {
    int i  = blockIdx.x * 256 + threadIdx.x;
    int sp = blockIdx.y;                      // 0..15
    if (i < FS) {
        int s = sp >> 1, plane = sp & 1;
        int src = L - 1 - s - i;
        float v = (src >= 0 && src < L) ? filt[src] : 0.0f;
        u16_t h, l;
        split_bf16(v, h, l);
        frws[(size_t)sp * (size_t)FS + i] = plane ? l : h;
    }
}

// ---------------- WMMA tile step (aligned-copy scheme) ----------------------
// A (16x32 bf16 MxK): row M=l15; quads at elem l15*XP + kh*32 + hl*16 (+8).
// B (32x16 bf16 KxN): elem i -> K=hl*16+i, N=l15; B[k][n]=filt[n-k] =
//   fr[bb+i], bb = 255 - (w*32+nt*16+l15) + kh*32 + hl*16.
//   bb mod 8 = (7-l15)&7 = s  ->  read copy s at idx = bb - s (mult of 8).

__device__ __forceinline__ void
mma_step_copies(const u16_t* xsh, const u16_t* xsl, const u16_t* frc,
                int hl, int l15, int w, v8f c[2]) {
    const int s7 = (7 - l15) & 7;
    const u16_t* fhb = frc + (size_t)(s7 * 2) * FCP;
    const u16_t* flb = fhb + FCP;

    #pragma unroll
    for (int kh = 0; kh < 2; ++kh) {
        const int aoff = l15 * XP + kh * 32 + hl * 16;   // byte 144*l15+64*kh+32*hl
        FragB16 ah, al;
        ld2q(ah, &xsh[aoff], &xsh[aoff + 8]);
        ld2q(al, &xsl[aoff], &xsl[aoff + 8]);

        #pragma unroll
        for (int nt = 0; nt < 2; ++nt) {
            const int idx = 255 - (w * 32 + nt * 16 + l15) + kh * 32 + hl * 16 - s7;
            FragB16 bh, bl;
            ld2q(bh, &fhb[idx], &fhb[idx + 8]);
            ld2q(bl, &flb[idx], &flb[idx + 8]);

            c[nt] = __builtin_amdgcn_wmma_f32_16x16x32_bf16(false, ah.v, false, bh.v,
                                                            (short)0, c[nt], false, false);
            c[nt] = __builtin_amdgcn_wmma_f32_16x16x32_bf16(false, ah.v, false, bl.v,
                                                            (short)0, c[nt], false, false);
            c[nt] = __builtin_amdgcn_wmma_f32_16x16x32_bf16(false, al.v, false, bh.v,
                                                            (short)0, c[nt], false, false);
        }
    }
}

// ---------------- main GEMM kernel (pre-converted bf16 planes) --------------

__global__ void __launch_bounds__(256)
futurefill_wmma_pre(const u16_t* __restrict__ xh, const u16_t* __restrict__ xl,
                    const u16_t* __restrict__ frws, float* __restrict__ out,
                    int T, int L, int FS) {
    const int n0 = blockIdx.x * BN;
    const int b0 = blockIdx.y * 16;
    const int tid  = threadIdx.x;
    const int w    = tid >> 5;
    const int lane = tid & 31;
    const int hl   = lane >> 4;
    const int l15  = lane & 15;

    __shared__ u16_t stage[2 * STG];          // double-buffered staged region
    // layout per buffer: [0, 16*XP) xs_hi | [16*XP, 32*XP) xs_lo
    //                    [32*XP, 32*XP + 16*FCP) 16 filter copy-planes

    v8f c[2] = {};

    // ---- per-thread staging geometry (one 16B chunk per async issue) ----
    // x tile: 2 planes x 16 rows x 64 cols = 256 chunks.
    const int plane = tid >> 7;
    const int xt    = tid & 127;
    const int xr    = xt >> 3;
    const int xc    = (xt & 7) * 8;
    const int xoff  = plane * 16 * XP + xr * XP + xc;   // elem offset in buffer
    const u16_t* xsrc = (plane ? xl : xh) + (size_t)(b0 + xr) * (size_t)T + xc;

    // filter: 16 copy-planes x 40 chunks = 640 chunks; thread handles
    // chunks tid, tid+256, and (tid<128) tid+512.
    int f, sp, jj;
    f = tid;       sp = f / 40; jj = (f % 40) * 8;
    const int f0off = 32 * XP + sp * FCP + jj;
    const u16_t* fg0 = frws + (size_t)sp * (size_t)FS + jj;
    f = tid + 256; sp = f / 40; jj = (f % 40) * 8;
    const int f1off = 32 * XP + sp * FCP + jj;
    const u16_t* fg1 = frws + (size_t)sp * (size_t)FS + jj;
    const bool has2 = (tid < 128);            // wave-uniform (waves 0-3)
    int f2off = f0off;
    const u16_t* fg2 = fg0;
    if (has2) {
        f = tid + 512; sp = f / 40; jj = (f % 40) * 8;
        f2off = 32 * XP + sp * FCP + jj;
        fg2 = frws + (size_t)sp * (size_t)FS + jj;
    }

    // issue all async chunks for lag-block k0 into stage buffer `buf`
    auto issue = [&](int buf, int k0) {
        const int bo = buf * STG;                       // elem offset of buffer
        const int fbase = L - 256 - n0 + k0;            // mult of 8 -> 16B aligned
#if USE_ASYNC_STAGE
        const unsigned lx = (unsigned)(size_t)(void*)&stage[bo + xoff];
        const u16_t* gx = xsrc + k0;
        asm volatile("global_load_async_to_lds_b128 %0, %1, off"
                     :: "v"(lx), "v"(gx) : "memory");
        const unsigned l0 = (unsigned)(size_t)(void*)&stage[bo + f0off];
        const u16_t* g0 = fg0 + fbase;
        asm volatile("global_load_async_to_lds_b128 %0, %1, off"
                     :: "v"(l0), "v"(g0) : "memory");
        const unsigned l1 = (unsigned)(size_t)(void*)&stage[bo + f1off];
        const u16_t* g1 = fg1 + fbase;
        asm volatile("global_load_async_to_lds_b128 %0, %1, off"
                     :: "v"(l1), "v"(g1) : "memory");
        if (has2) {
            const unsigned l2 = (unsigned)(size_t)(void*)&stage[bo + f2off];
            const u16_t* g2 = fg2 + fbase;
            asm volatile("global_load_async_to_lds_b128 %0, %1, off"
                         :: "v"(l2), "v"(g2) : "memory");
        }
#else
        *(uint4*)&stage[bo + xoff]  = *(const uint4*)(xsrc + k0);
        *(uint4*)&stage[bo + f0off] = *(const uint4*)(fg0 + fbase);
        *(uint4*)&stage[bo + f1off] = *(const uint4*)(fg1 + fbase);
        if (has2)
            *(uint4*)&stage[bo + f2off] = *(const uint4*)(fg2 + fbase);
#endif
    };

    const int kend = n0 + BN;                 // causal diagonal for this block

    issue(0, 0);                              // prologue
    int par = 0;
    for (int k0 = 0; k0 < kend; k0 += BK) {
#if USE_ASYNC_STAGE
        asm volatile("s_wait_asynccnt 0x0" ::: "memory");   // tile k0 loaded
#endif
        __syncthreads();                      // everyone done reading buf[par^1]
        if (k0 + BK < kend) issue(par ^ 1, k0 + BK);        // prefetch next tile

        const u16_t* sb = &stage[par * STG];
        mma_step_copies(sb, sb + 16 * XP, sb + 32 * XP, hl, l15, w, c);
        par ^= 1;
    }

    // ---- store D: element r -> out[b0 + r + 8*hl, n0 + w*32 + nt*16 + l15] ----
    #pragma unroll
    for (int nt = 0; nt < 2; ++nt) {
        float* op = out + (size_t)(b0 + 8 * hl) * (size_t)T
                        + (size_t)(n0 + w * 32 + nt * 16 + l15);
        #pragma unroll
        for (int r = 0; r < 8; ++r) op[(size_t)r * (size_t)T] = c[nt][r];
    }
}

// ---------------- fallback: in-kernel conversion (ws too small) -------------

__device__ __forceinline__ void
mma_step_basic(const u16_t* xsh, const u16_t* xsl,
               const u16_t* frh, const u16_t* frl,
               int hl, int l15, int w, v8f c[2]) {
    #pragma unroll
    for (int kh = 0; kh < 2; ++kh) {
        const int aoff = l15 * XP + kh * 32 + hl * 16;
        FragB16 ah, al;
        ld2q(ah, &xsh[aoff], &xsh[aoff + 8]);
        ld2q(al, &xsl[aoff], &xsl[aoff + 8]);
        #pragma unroll
        for (int nt = 0; nt < 2; ++nt) {
            FragB16 bh, bl;
            const int bb = 255 - (w * 32 + nt * 16 + l15) + kh * 32 + hl * 16;
            #pragma unroll
            for (int i = 0; i < 16; ++i) {
                bh.s[i] = frh[bb + i];
                bl.s[i] = frl[bb + i];
            }
            c[nt] = __builtin_amdgcn_wmma_f32_16x16x32_bf16(false, ah.v, false, bh.v,
                                                            (short)0, c[nt], false, false);
            c[nt] = __builtin_amdgcn_wmma_f32_16x16x32_bf16(false, ah.v, false, bl.v,
                                                            (short)0, c[nt], false, false);
            c[nt] = __builtin_amdgcn_wmma_f32_16x16x32_bf16(false, al.v, false, bh.v,
                                                            (short)0, c[nt], false, false);
        }
    }
}

__global__ void __launch_bounds__(256)
futurefill_wmma_conv(const float* __restrict__ x, const float* __restrict__ filt,
                     float* __restrict__ out, int T, int L) {
    const int n0 = blockIdx.x * BN;
    const int b0 = blockIdx.y * 16;
    const int tid  = threadIdx.x;
    const int w    = tid >> 5;
    const int lane = tid & 31;
    const int hl   = lane >> 4;
    const int l15  = lane & 15;

    __shared__ u16_t xs_hi[16 * XP];
    __shared__ u16_t xs_lo[16 * XP];
    __shared__ u16_t fr_hi[FRW + 16];
    __shared__ u16_t fr_lo[FRW + 16];

    v8f c[2] = {};

    const int xr = tid >> 4;
    const int c4 = (tid & 15) * 4;

    const int kend = n0 + BN;
    for (int k0 = 0; k0 < kend; k0 += BK) {
        __syncthreads();
        {
            const float4 v4 = *(const float4*)&x[(size_t)(b0 + xr) * (size_t)T
                                                 + (size_t)(k0 + c4)];
            u16_t h0, l0, h1, l1, h2, l2, h3, l3;
            split_bf16(v4.x, h0, l0); split_bf16(v4.y, h1, l1);
            split_bf16(v4.z, h2, l2); split_bf16(v4.w, h3, l3);
            *(u32_t*)&xs_hi[xr * XP + c4]     = (u32_t)h0 | ((u32_t)h1 << 16);
            *(u32_t*)&xs_hi[xr * XP + c4 + 2] = (u32_t)h2 | ((u32_t)h3 << 16);
            *(u32_t*)&xs_lo[xr * XP + c4]     = (u32_t)l0 | ((u32_t)l1 << 16);
            *(u32_t*)&xs_lo[xr * XP + c4 + 2] = (u32_t)l2 | ((u32_t)l3 << 16);
        }
        #pragma unroll
        for (int j = tid; j < FRW; j += 256) {
            const int d = n0 - k0 + 255 - j;
            float v = (d >= 0 && d < L) ? filt[d] : 0.0f;
            u16_t h, l;
            split_bf16(v, h, l);
            fr_hi[j] = h; fr_lo[j] = l;
        }
        __syncthreads();

        mma_step_basic(xs_hi, xs_lo, fr_hi, fr_lo, hl, l15, w, c);
    }

    #pragma unroll
    for (int nt = 0; nt < 2; ++nt) {
        float* op = out + (size_t)(b0 + 8 * hl) * (size_t)T
                        + (size_t)(n0 + w * 32 + nt * 16 + l15);
        #pragma unroll
        for (int r = 0; r < 8; ++r) op[(size_t)r * (size_t)T] = c[nt][r];
    }
}

// ---------------- host launcher ---------------------------------------------

extern "C" void kernel_launch(void* const* d_in, const int* in_sizes, int n_in,
                              void* d_out, int out_size, void* d_ws, size_t ws_size,
                              hipStream_t stream) {
    const float* x    = (const float*)d_in[0];
    const float* filt = (const float*)d_in[1];
    float* out        = (float*)d_out;

    const int L = in_sizes[1];              // 32768
    const int T = L;                        // this problem has T == L
    const int B = in_sizes[0] / T;          // 256
    const size_t BT = (size_t)B * (size_t)T;
    const int FS = L + FCP;                 // per copy-plane extent in ws

    const size_t need = 4 * BT + 32 * (size_t)FS;   // bytes
    dim3 grid(T / BN, B / 16);              // (128, 16)

    if (ws_size >= need) {
        u16_t* xh   = (u16_t*)d_ws;
        u16_t* xl   = xh + BT;
        u16_t* frws = xl + BT;              // 16 copy-planes, each FS elems
        prep_split_x<<<(unsigned)((BT + 255) / 256), 256, 0, stream>>>(x, xh, xl, (int)BT);
        dim3 fgrid((unsigned)((FS + 255) / 256), 16);
        prep_split_filt_copies<<<fgrid, 256, 0, stream>>>(filt, frws, L, FS);
        futurefill_wmma_pre<<<grid, 256, 0, stream>>>(xh, xl, frws, out, T, L, FS);
    } else {
        futurefill_wmma_conv<<<grid, 256, 0, stream>>>(x, filt, out, T, L);
    }
}
